// VQVAE_76338748719693
// MI455X (gfx1250) — matrix operand order repaired and checked by reading, not compiled, not run on hardware
//
#include <hip/hip_runtime.h>
#include <hip/hip_bf16.h>
#include <math.h>

// ---------------------------------------------------------------------------
// Types for CDNA5 WMMA
// ---------------------------------------------------------------------------
typedef __bf16 v16bf __attribute__((ext_vector_type(16)));
typedef float  v8f   __attribute__((ext_vector_type(8)));

union Frag {
    unsigned int u[8];
    v16bf        v;
};

__device__ __forceinline__ v8f wmma_bf16(const Frag& a, const Frag& b, v8f c) {
    // v_wmma_f32_16x16x32_bf16  (neg_a, A, neg_b, B, c_mod, C, reuse_a, reuse_b)
    return __builtin_amdgcn_wmma_f32_16x16x32_bf16(false, a.v, false, b.v,
                                                   (short)0, c, false, false);
}

__device__ __forceinline__ unsigned short f2bf(float f) {
    unsigned int u = __float_as_uint(f);
    unsigned int r = (u + 0x7FFFu + ((u >> 16) & 1u)) >> 16;   // RNE
    return (unsigned short)r;
}
__device__ __forceinline__ float bf2f(unsigned short h) {
    return __uint_as_float(((unsigned int)h) << 16);
}
__device__ __forceinline__ float gelu_tanh(float x) {
    float x3 = x * x * x;
    return 0.5f * x * (1.0f + tanhf(0.7978845608028654f * (x + 0.044715f * x3)));
}

// A-fragment dword index within a 16-dword (32 bf16) row (ISA 16-bit A layout)
__device__ __forceinline__ int a_dw(int v, int hi) {
    return (v & 3) + (hi ? 4 : 0) + ((v >> 2) << 3);
}
// B-fragment dword index within a 16-dword row of B^T (n-major) storage
__device__ __forceinline__ int b_dw(int v, int hi) {
    return (v & 3) + (hi ? 4 : 0) + ((v >> 2) << 3);
}

// LDS tile row stride in dwords: multiple of 4 (16-B alignment for b128 LDS
// traffic) and 20 mod 64 banks -> 16 lanes with stride-20 rows hit 16 distinct
// banks on fragment reads.
#define LROW 20

// ---------------------------------------------------------------------------
// Problem constants
// ---------------------------------------------------------------------------
#define BB   16
#define HH   256
#define WW   256
#define HID  128
#define EMB  64
#define KCB  1024
#define NPIX (BB * 64 * 64)          // 65536 latent pixels

// Workspace layout (bytes). Requires ws_size >= ~363 MB.
static constexpr size_t OFF_H1   = 0;                              // 33.5M bf16  (reused as h3)
static constexpr size_t OFF_H2   = OFF_H1  + 67108864ull;          // 8.39M bf16
static constexpr size_t OFF_ZE   = OFF_H2  + 16777216ull;          // 4.19M f32
static constexpr size_t OFF_ZQ   = OFF_ZE  + 16777216ull;          // 4.19M bf16
static constexpr size_t OFF_G2   = OFF_ZQ  + 8388608ull;           // 134M bf16
static constexpr size_t OFF_W2T  = OFF_G2  + 268435456ull;         // 128x2048 bf16
static constexpr size_t OFF_W3T  = OFF_W2T + 524288ull;            // 64x128 bf16
static constexpr size_t OFF_CB   = OFF_W3T + 16384ull;             // 1024x64 bf16
static constexpr size_t OFF_E2   = OFF_CB  + 131072ull;            // 1024 f32
static constexpr size_t OFF_D1T  = OFF_E2  + 4096ull;              // 4x128x256 bf16
static constexpr size_t OFF_D2T  = OFF_D1T + 262144ull;            // 4x128x512 bf16
static constexpr size_t OFF_IDX  = OFF_D2T + 524288ull;            // 65536 i32
static constexpr size_t OFF_PART = OFF_IDX + 262144ull;            // 256 f32

// ---------------------------------------------------------------------------
// Weight prep: fp32 -> bf16, transposed to [N][K] GEMM layout.
// blockIdx.y selects section.
// ---------------------------------------------------------------------------
__global__ void prep_weights(const float* __restrict__ enc_w2,
                             const float* __restrict__ enc_w3,
                             const float* __restrict__ codebook,
                             const float* __restrict__ dec_w1,
                             const float* __restrict__ dec_w2,
                             unsigned short* __restrict__ w2T,
                             unsigned short* __restrict__ w3T,
                             unsigned short* __restrict__ cbT,
                             float* __restrict__ e2,
                             unsigned short* __restrict__ d1T,
                             unsigned short* __restrict__ d2T) {
    int id = blockIdx.x * 256 + threadIdx.x;
    int sec = blockIdx.y;
    if (sec == 0) {                       // enc_w2 [4,4,128,128] -> w2T[n][ (kh4+kw)*128+c ]
        if (id >= 128 * 2048) return;
        int n = id >> 11, k = id & 2047;
        w2T[id] = f2bf(enc_w2[(size_t)k * 128 + n]);
    } else if (sec == 1) {                // enc_w3 [1,1,128,64] -> w3T[n][c]
        if (id >= 64 * 128) return;
        int n = id >> 7, c = id & 127;
        w3T[id] = f2bf(enc_w3[c * 64 + n]);
    } else if (sec == 2) {                // codebook fp32 -> bf16 (row-major [1024][64])
        if (id >= KCB * EMB) return;
        cbT[id] = f2bf(codebook[id]);
    } else if (sec == 3) {                // e2[n] = sum_k cb[n][k]^2
        if (id >= KCB) return;
        float s = 0.f;
        for (int k = 0; k < EMB; ++k) { float c = codebook[id * EMB + k]; s += c * c; }
        e2[id] = s;
    } else if (sec == 4) {                // dec_w1 [4,4,64,128] -> d1T[p][n][ (th2+tw)*64+c ]
        if (id >= 4 * 128 * 256) return;
        int p = id >> 15, rem = id & 32767;
        int n = rem >> 8, k = rem & 255;
        int py = p >> 1, px = p & 1;
        int th = (k >> 7) & 1, tw = (k >> 6) & 1, c = k & 63;
        int kh = 2 * th + py, kw = 2 * tw + px;
        d1T[id] = f2bf(dec_w1[(((kh * 4 + kw) * 64) + c) * 128 + n]);
    } else {                              // dec_w2 [4,4,128,128] -> d2T[p][n][ (th2+tw)*128+c ]
        if (id >= 4 * 128 * 512) return;
        int p = id >> 16, rem = id & 65535;
        int n = rem >> 9, k = rem & 511;
        int py = p >> 1, px = p & 1;
        int th = (k >> 8) & 1, tw = (k >> 7) & 1, c = k & 127;
        int kh = 2 * th + py, kw = 2 * tw + px;
        d2T[id] = f2bf(dec_w2[(((kh * 4 + kw) * 128) + c) * 128 + n]);
    }
}

// ---------------------------------------------------------------------------
// Encoder conv1: 4x4 s2 SAME, C=3 -> 128, GELU, bf16 out. K=48 too small for
// WMMA; scalar FMA, weights cached in LDS. 16 pixels / block, 8 ch / thread.
// ---------------------------------------------------------------------------
__global__ void conv1_kernel(const float* __restrict__ x,
                             const float* __restrict__ w,
                             const float* __restrict__ bias,
                             unsigned short* __restrict__ h1) {
    __shared__ float wl[4 * 4 * 3 * 128];
    __shared__ float bl[128];
    int tid = threadIdx.x;
    for (int i = tid; i < 6144; i += 256) wl[i] = w[i];
    if (tid < 128) bl[tid] = bias[tid];
    __syncthreads();

    int pix = blockIdx.x * 16 + (tid >> 4);
    int cg  = (tid & 15) * 8;
    int b = pix >> 14, r = pix & 16383, oy = r >> 7, ox = r & 127;

    float acc[8];
    #pragma unroll
    for (int j = 0; j < 8; ++j) acc[j] = bl[cg + j];

    for (int kh = 0; kh < 4; ++kh) {
        int iy = 2 * oy - 1 + kh;
        if (iy < 0 || iy >= HH) continue;
        for (int kw = 0; kw < 4; ++kw) {
            int ix = 2 * ox - 1 + kw;
            if (ix < 0 || ix >= WW) continue;
            const float* xp = x + ((size_t)(b * HH + iy) * WW + ix) * 3;
            #pragma unroll
            for (int cc = 0; cc < 3; ++cc) {
                float xv = xp[cc];
                const float* wp = &wl[((kh * 4 + kw) * 3 + cc) * 128 + cg];
                #pragma unroll
                for (int j = 0; j < 8; ++j) acc[j] += xv * wp[j];
            }
        }
    }
    unsigned short* out = h1 + (size_t)pix * 128 + cg;
    #pragma unroll
    for (int j = 0; j < 8; ++j) out[j] = f2bf(gelu_tanh(acc[j]));
}

// ---------------------------------------------------------------------------
// Implicit GEMM with bf16 WMMA, f32 accumulate, double-buffered LDS.
// B tiles (weights, always valid) stream via GLOBAL_LOAD_ASYNC_TO_LDS_B128
// (ASYNCcnt / s_wait_asynccnt); A tiles (im2col, needs zero-fill) are
// software-pipelined through VGPRs so their global latency hides behind WMMA.
// MODE 0: enc conv2  (A=h1 im2col, K=2048, N=128, GELU, bf16 out)
// MODE 1: enc 1x1    (A=h2,        K=128,  N=64,  bias only, f32 out z_e)
// MODE 2: dec convT1 (A=zq gather, K=256,  N=128, GELU, bf16 out, 4 parities)
// MODE 3: dec convT2 (A=h3 gather, K=512,  N=128, GELU, bf16 out, 4 parities)
// 256 threads = 8 waves; tile M=128 x N.
// ---------------------------------------------------------------------------
template <int MODE> struct GP;
template <> struct GP<0> { static constexpr int K = 2048, N = 128; };
template <> struct GP<1> { static constexpr int K = 128,  N = 64;  };
template <> struct GP<2> { static constexpr int K = 256,  N = 128; };
template <> struct GP<3> { static constexpr int K = 512,  N = 128; };

template <int MODE>
__global__ void gemm_kernel(const unsigned short* __restrict__ A,
                            const unsigned short* __restrict__ Bw,
                            const float* __restrict__ bias,
                            void* __restrict__ outp) {
    constexpr int K   = GP<MODE>::K;
    constexpr int N   = GP<MODE>::N;
    constexpr int MF  = (N == 128) ? 2 : 1;
    constexpr int NF  = 4;
    constexpr int APT = (N == 128) ? 2 : 1;   // async b128 per thread per B tile

    __shared__ unsigned int lA[2][128 * LROW];
    __shared__ unsigned int lB[2][128 * LROW];

    const int tid  = threadIdx.x;
    const int wave = tid >> 5;
    const int lane = tid & 31;
    const int lm   = lane & 15;
    const int hi   = lane >> 4;

    int wm, wn;
    if constexpr (N == 128) { wm = wave & 3; wn = wave >> 2; }
    else                    { wm = wave;     wn = 0;         }
    const int mrow0 = wm * (MF * 16);
    const int ncol0 = wn * 64;

    int py = 0, px = 0;
    if constexpr (MODE == 2 || MODE == 3) {
        py = (int)(blockIdx.y >> 1);
        px = (int)(blockIdx.y & 1);
        Bw += (size_t)blockIdx.y * N * K;   // per-parity weight slice
    }

    const unsigned int* A32 = (const unsigned int*)A;
    const int r_ld = tid >> 1, h_ld = tid & 1;

    // ---- A tile: im2col gather of 32 contiguous-channel bf16 into VGPRs
    auto load_A_regs = [&](int kb, unsigned int av[8]) {
        int gm = blockIdx.x * 128 + r_ld;
        long base = 0; bool ok = true;
        if constexpr (MODE == 0) {
            int kc = kb >> 5, kh = kc >> 4, kw = (kc >> 2) & 3, c0 = (kc & 3) << 5;
            int b = gm >> 12, rr = gm & 4095, oy = rr >> 6, ox = rr & 63;
            int iy = 2 * oy - 1 + kh, ix = 2 * ox - 1 + kw;
            ok = (iy >= 0) & (iy < 128) & (ix >= 0) & (ix < 128);
            base = (((long)((b * 128 + iy) * 128 + ix)) << 7) + c0;
        } else if constexpr (MODE == 1) {
            base = (long)gm * 128 + kb;
        } else if constexpr (MODE == 2) {
            int kc = kb >> 5, th = kc >> 2, tw = (kc >> 1) & 1, c0 = (kc & 1) << 5;
            int b = gm >> 12, rr = gm & 4095, oy2 = rr >> 6, ox2 = rr & 63;
            int iy = oy2 + th - 1 + py, ix = ox2 + tw - 1 + px;
            ok = (iy >= 0) & (iy < 64) & (ix >= 0) & (ix < 64);
            base = (((long)((b * 64 + iy) * 64 + ix)) << 6) + c0;
        } else {
            int kc = kb >> 5, th = kc >> 3, tw = (kc >> 2) & 1, c0 = (kc & 3) << 5;
            int b = gm >> 14, rr = gm & 16383, oy2 = rr >> 7, ox2 = rr & 127;
            int iy = oy2 + th - 1 + py, ix = ox2 + tw - 1 + px;
            ok = (iy >= 0) & (iy < 128) & (ix >= 0) & (ix < 128);
            base = (((long)((b * 128 + iy) * 128 + ix)) << 7) + c0;
        }
        const unsigned int* s = A32 + (base >> 1) + h_ld * 8;
        #pragma unroll
        for (int j = 0; j < 8; ++j) av[j] = ok ? s[j] : 0u;
    };
    auto store_A_lds = [&](int buf, const unsigned int av[8]) {
        unsigned int* d = &lA[buf][r_ld * LROW + h_ld * 8];
        #pragma unroll
        for (int j = 0; j < 8; ++j) d[j] = av[j];
    };
    // ---- B tile: weights, always in-bounds -> async global->LDS (ASYNCcnt)
    auto issue_B_async = [&](int buf, int kb) {
        #pragma unroll
        for (int i = 0; i < APT; ++i) {
            int id = tid * APT + i;
            int n = id >> 2, q = id & 3;
            unsigned int lds_off =
                (unsigned int)(size_t)&lB[buf][n * LROW + q * 4];
            unsigned long long ga =
                (unsigned long long)(size_t)(Bw + (size_t)n * K + kb + q * 8);
            asm volatile("global_load_async_to_lds_b128 %0, %1, off"
                         :: "v"(lds_off), "v"(ga)
                         : "memory");
        }
    };

    v8f acc[MF][NF];
    #pragma unroll
    for (int mi = 0; mi < MF; ++mi)
        #pragma unroll
        for (int ni = 0; ni < NF; ++ni)
            acc[mi][ni] = (v8f){0.f, 0.f, 0.f, 0.f, 0.f, 0.f, 0.f, 0.f};

    // ---- prologue: stage tile 0
    unsigned int av[8];
    load_A_regs(0, av);
    issue_B_async(0, 0);
    store_A_lds(0, av);
    int cur = 0;

    for (int kb = 0; kb < K; kb += 32) {
        const bool more = (kb + 32) < K;
        if (more) {
            load_A_regs(kb + 32, av);       // next A -> VGPRs (latency hidden)
            issue_B_async(cur ^ 1, kb + 32);
        }
        // Drain current tile's async B loads (uniform APT outstanding for the
        // next tile remain in flight), then make LDS visible to all waves.
        if (more) {
            if constexpr (APT == 2) asm volatile("s_wait_asynccnt 0x2" ::: "memory");
            else                    asm volatile("s_wait_asynccnt 0x1" ::: "memory");
        } else {
            asm volatile("s_wait_asynccnt 0x0" ::: "memory");
        }
        __syncthreads();

        // ---- fragments + WMMA on current buffer
        Frag af[MF], bfr[NF];
        #pragma unroll
        for (int mi = 0; mi < MF; ++mi) {
            int row = mrow0 + mi * 16 + lm;
            #pragma unroll
            for (int v = 0; v < 8; ++v) af[mi].u[v] = lA[cur][row * LROW + a_dw(v, hi)];
        }
        #pragma unroll
        for (int ni = 0; ni < NF; ++ni) {
            int n = ncol0 + ni * 16 + lm;
            #pragma unroll
            for (int v = 0; v < 8; ++v) bfr[ni].u[v] = lB[cur][n * LROW + b_dw(v, hi)];
        }
        #pragma unroll
        for (int mi = 0; mi < MF; ++mi)
            #pragma unroll
            for (int ni = 0; ni < NF; ++ni)
                acc[mi][ni] = wmma_bf16(af[mi], bfr[ni], acc[mi][ni]);

        if (more) store_A_lds(cur ^ 1, av); // commit next A after WMMAs
        __syncthreads();                    // cur buffer free for reuse
        cur ^= 1;
    }

    // ---- epilogue: bias (+GELU), store
    #pragma unroll
    for (int mi = 0; mi < MF; ++mi) {
        #pragma unroll
        for (int ni = 0; ni < NF; ++ni) {
            int n  = ncol0 + ni * 16 + lm;
            float bn = bias[n];
            #pragma unroll
            for (int v = 0; v < 8; ++v) {
                int ml = mrow0 + mi * 16 + v + hi * 8;
                int gm = blockIdx.x * 128 + ml;
                float val = acc[mi][ni][v] + bn;
                if constexpr (MODE == 1) {
                    ((float*)outp)[(size_t)gm * 64 + n] = val;
                } else {
                    val = gelu_tanh(val);
                    size_t addr;
                    if constexpr (MODE == 0) {
                        addr = (size_t)gm * 128 + n;
                    } else if constexpr (MODE == 2) {
                        int b = gm >> 12, rr = gm & 4095, oy2 = rr >> 6, ox2 = rr & 63;
                        int oy = 2 * oy2 + py, ox = 2 * ox2 + px;
                        addr = ((size_t)(b * 128 + oy) * 128 + ox) * 128 + n;
                    } else {
                        int b = gm >> 14, rr = gm & 16383, oy2 = rr >> 7, ox2 = rr & 127;
                        int oy = 2 * oy2 + py, ox = 2 * ox2 + px;
                        addr = ((size_t)(b * 256 + oy) * 256 + ox) * 128 + n;
                    }
                    ((unsigned short*)outp)[addr] = f2bf(val);
                }
            }
        }
    }
}

// ---------------------------------------------------------------------------
// VQ argmin: dist = z^2 + e2 - 2 z.e ; z^2 is row-constant -> argmin over
// (e2 - 2 dot). Each wave owns a 16-row A strip (K=64, resident in regs) and
// sweeps all 1024 codes in 16-col WMMA tiles.
// ---------------------------------------------------------------------------
__global__ void vq_argmin(const float* __restrict__ ze,
                          const unsigned short* __restrict__ cb,
                          const float* __restrict__ e2,
                          int* __restrict__ idxbuf) {
    const int tid  = threadIdx.x;
    const int wave = tid >> 5;
    const int lane = tid & 31;
    const int lm   = lane & 15;
    const int hi   = lane >> 4;
    const int m0   = blockIdx.x * 128 + wave * 16;
    const int m    = m0 + lm;

    // Build A fragments (2 x K32) from fp32 z_e, converting to bf16 pairs.
    Frag af[2];
    #pragma unroll
    for (int f = 0; f < 2; ++f)
        #pragma unroll
        for (int v = 0; v < 8; ++v) {
            int k0 = f * 32 + v * 2 + hi * 16;
            unsigned int lo = f2bf(ze[(size_t)m * 64 + k0]);
            unsigned int hi16 = f2bf(ze[(size_t)m * 64 + k0 + 1]);
            af[f].u[v] = lo | (hi16 << 16);
        }

    float best[8];
    int   bidx[8];
    #pragma unroll
    for (int v = 0; v < 8; ++v) { best[v] = 3.4e38f; bidx[v] = 0; }

    const unsigned int* cb32 = (const unsigned int*)cb;
    for (int nt = 0; nt < 64; ++nt) {
        float e2v = e2[nt * 16 + lm];
        Frag bfr[2];
        #pragma unroll
        for (int f = 0; f < 2; ++f) {
            int n = nt * 16 + lm;
            #pragma unroll
            for (int v = 0; v < 8; ++v)
                bfr[f].u[v] = cb32[(size_t)n * 32 + f * 16 + v + hi * 8];
        }
        v8f acc = (v8f){0.f, 0.f, 0.f, 0.f, 0.f, 0.f, 0.f, 0.f};
        acc = wmma_bf16(af[0], bfr[0], acc);
        acc = wmma_bf16(af[1], bfr[1], acc);
        int n = nt * 16 + lm;
        #pragma unroll
        for (int v = 0; v < 8; ++v) {
            float dist = e2v - 2.0f * acc[v];
            if (dist < best[v]) { best[v] = dist; bidx[v] = n; }
        }
    }

    // Cross-lane min within each 16-lane half (lanes hold distinct n, same m)
    #pragma unroll
    for (int v = 0; v < 8; ++v) {
        float bv = best[v];
        int   bi = bidx[v];
        #pragma unroll
        for (int off = 8; off >= 1; off >>= 1) {
            float ov = __shfl_xor(bv, off, 32);
            int   oi = __shfl_xor(bi, off, 32);
            if (ov < bv || (ov == bv && oi < bi)) { bv = ov; bi = oi; }
        }
        if (lm == 0) idxbuf[m0 + hi * 8 + v] = bi;
    }
}

// ---------------------------------------------------------------------------
// Gather z_q (bf16 for decoder), write tokens, partial commit-loss sums.
// ---------------------------------------------------------------------------
__global__ void gather_zq(const int* __restrict__ idxbuf,
                          const float* __restrict__ codebook,
                          const float* __restrict__ ze,
                          unsigned short* __restrict__ zq,
                          float* __restrict__ partials,
                          float* __restrict__ tok_out) {
    __shared__ float red[256];
    int m = blockIdx.x * 256 + threadIdx.x;
    int i = idxbuf[m];
    float s = 0.f;
    for (int k = 0; k < EMB; ++k) {
        float c = codebook[(size_t)i * EMB + k];
        float z = ze[(size_t)m * EMB + k];
        float d = c - z;
        s += d * d;
        zq[(size_t)m * EMB + k] = f2bf(c);
    }
    tok_out[m] = (float)i;
    red[threadIdx.x] = s;
    __syncthreads();
    for (int st = 128; st > 0; st >>= 1) {
        if (threadIdx.x < st) red[threadIdx.x] += red[threadIdx.x + st];
        __syncthreads();
    }
    if (threadIdx.x == 0) partials[blockIdx.x] = red[0];
}

__global__ void reduce_loss(const float* __restrict__ partials,
                            float* __restrict__ loss_out) {
    __shared__ float red[256];
    red[threadIdx.x] = partials[threadIdx.x];
    __syncthreads();
    for (int st = 128; st > 0; st >>= 1) {
        if (threadIdx.x < st) red[threadIdx.x] += red[threadIdx.x + st];
        __syncthreads();
    }
    if (threadIdx.x == 0) {
        float v = red[0] / (float)((size_t)NPIX * EMB);
        loss_out[0] = v;   // commit
        loss_out[1] = v;   // codebook loss (identical in forward)
    }
}

// ---------------------------------------------------------------------------
// Final 1x1 conv: 128 -> 3, fp32 output.
// ---------------------------------------------------------------------------
__global__ void conv_out(const unsigned short* __restrict__ g2,
                         const float* __restrict__ w3,
                         const float* __restrict__ b3,
                         float* __restrict__ out) {
    __shared__ float wl[384];
    int tid = threadIdx.x;
    for (int i = tid; i < 384; i += 256) wl[i] = w3[i];
    __syncthreads();
    int m = blockIdx.x * 256 + tid;
    float a0 = b3[0], a1 = b3[1], a2 = b3[2];
    const unsigned short* gp = g2 + (size_t)m * 128;
    for (int k = 0; k < 128; ++k) {
        float g = bf2f(gp[k]);
        a0 += g * wl[k * 3 + 0];
        a1 += g * wl[k * 3 + 1];
        a2 += g * wl[k * 3 + 2];
    }
    float* op = out + (size_t)m * 3;
    op[0] = a0; op[1] = a1; op[2] = a2;
}

// ---------------------------------------------------------------------------
// Launcher
// ---------------------------------------------------------------------------
extern "C" void kernel_launch(void* const* d_in, const int* in_sizes, int n_in,
                              void* d_out, int out_size, void* d_ws, size_t ws_size,
                              hipStream_t stream) {
    (void)in_sizes; (void)n_in; (void)out_size; (void)ws_size;

    const float* x        = (const float*)d_in[0];
    const float* enc_w1   = (const float*)d_in[1];
    const float* enc_b1   = (const float*)d_in[2];
    const float* enc_w2   = (const float*)d_in[3];
    const float* enc_b2   = (const float*)d_in[4];
    const float* enc_w3   = (const float*)d_in[5];
    const float* enc_b3   = (const float*)d_in[6];
    const float* codebook = (const float*)d_in[7];
    const float* dec_w1   = (const float*)d_in[8];
    const float* dec_b1   = (const float*)d_in[9];
    const float* dec_w2   = (const float*)d_in[10];
    const float* dec_b2   = (const float*)d_in[11];
    const float* dec_w3   = (const float*)d_in[12];
    const float* dec_b3   = (const float*)d_in[13];

    char* ws = (char*)d_ws;
    unsigned short* h1   = (unsigned short*)(ws + OFF_H1);   // also h3
    unsigned short* h2   = (unsigned short*)(ws + OFF_H2);
    float*          ze   = (float*)         (ws + OFF_ZE);
    unsigned short* zq   = (unsigned short*)(ws + OFF_ZQ);
    unsigned short* g2   = (unsigned short*)(ws + OFF_G2);
    unsigned short* w2T  = (unsigned short*)(ws + OFF_W2T);
    unsigned short* w3T  = (unsigned short*)(ws + OFF_W3T);
    unsigned short* cbT  = (unsigned short*)(ws + OFF_CB);
    float*          e2   = (float*)         (ws + OFF_E2);
    unsigned short* d1T  = (unsigned short*)(ws + OFF_D1T);
    unsigned short* d2T  = (unsigned short*)(ws + OFF_D2T);
    int*            idxb = (int*)           (ws + OFF_IDX);
    float*          part = (float*)         (ws + OFF_PART);

    float* xrec = (float*)d_out;                       // 3,145,728
    float* tok  = xrec + (size_t)BB * HH * WW * 3;     // 65,536
    float* loss = tok + NPIX;                          // 2

    // 0) weights -> bf16 [N][K] (+parity slices), e2
    prep_weights<<<dim3(1024, 6), 256, 0, stream>>>(enc_w2, enc_w3, codebook,
                                                    dec_w1, dec_w2,
                                                    w2T, w3T, cbT, e2, d1T, d2T);
    // 1) enc conv1 (scalar, K=48)
    conv1_kernel<<<16384, 256, 0, stream>>>(x, enc_w1, enc_b1, h1);
    // 2) enc conv2: M=65536, K=2048, N=128
    gemm_kernel<0><<<dim3(512, 1), 256, 0, stream>>>(h1, w2T, enc_b2, (void*)h2);
    // 3) enc 1x1 -> z_e fp32: M=65536, K=128, N=64
    gemm_kernel<1><<<dim3(512, 1), 256, 0, stream>>>(h2, w3T, enc_b3, (void*)ze);
    // 4) VQ argmin over 1024 codes
    vq_argmin<<<512, 256, 0, stream>>>(ze, cbT, e2, idxb);
    // 5) gather z_q, tokens, loss partials; reduce
    gather_zq<<<256, 256, 0, stream>>>(idxb, codebook, ze, zq, part, tok);
    reduce_loss<<<1, 256, 0, stream>>>(part, loss);
    // 6) dec convT1: 4 parity GEMMs, M=65536 each, K=256, N=128 (h3 reuses h1)
    gemm_kernel<2><<<dim3(512, 4), 256, 0, stream>>>(zq, d1T, dec_b1, (void*)h1);
    // 7) dec convT2: 4 parity GEMMs, M=262144 each, K=512, N=128
    gemm_kernel<3><<<dim3(2048, 4), 256, 0, stream>>>(h1, d2T, dec_b2, (void*)g2);
    // 8) final 1x1 -> x_rec fp32
    conv_out<<<4096, 256, 0, stream>>>(g2, dec_w3, dec_b3, xrec);
}
